// RVQVAE_2D_79963701117555
// MI455X (gfx1250) — compile-verified, matrix operand order checked by reading, not run on hardware
//
#include <hip/hip_runtime.h>

// ============================================================================
// RVQ-VAE (1D + 2D branches) for MI455X / gfx1250.
//
// Every conv / linear is an implicit GEMM on v_wmma_f32_16x16x32_f16:
//   - 128-thread blocks (4 waves); the 4 waves cover 64 output channels and
//     share one 16-position im2col B tile staged in LDS as f16
//     (40-half row stride -> 16B-aligned ds_load_b128 fragment reads).
//   - Weights (A) fast path: K multiple of 32 -> 4x b128 global loads/lane.
//   - All bounds handling is clamp+select (branch-free): loads are always
//     issued from clamped in-bounds addresses and the *value* is masked,
//     so no exec toggling / saveexec branches in any inner loop.
//   - Kernel size is a template parameter so index math is shifts/mads.
// RVQ / transposes / upsamples are cheap elementwise or block-reduce kernels.
// ============================================================================

typedef __attribute__((ext_vector_type(16))) _Float16 v16h;
typedef __attribute__((ext_vector_type(8)))  _Float16 v8h;
typedef __attribute__((ext_vector_type(8)))  float    v8f;
typedef __attribute__((ext_vector_type(4)))  float    v4f;

static inline int cdiv(int a, int b) { return (a + b - 1) / b; }

// ---------------------------------------------------------------------------
// WMMA implicit-GEMM 1D convolution (templated on kernel size).
// x: (N, Cin, Tin)  w: (Cout, Cin, KS)  y: (N, Cout, Tout)
// ---------------------------------------------------------------------------
template <int KS>
__global__ __launch_bounds__(128)
void k_conv1d_wmma(const float* __restrict__ x,
                   const float* __restrict__ w,
                   const float* __restrict__ bias,
                   const float* __restrict__ res,
                   float* __restrict__ y,
                   int N, int Cin, int Tin, int Cout, int Tout,
                   int st, int pad, int dil, int relu_in, int relu_out)
{
    const int Ktot = Cin * KS;
    const int Npos = N * Tout;
    const int tid  = threadIdx.x;
    const int lane = tid & 31;
    const int wv   = tid >> 5;
    const int m0   = blockIdx.y * 64 + wv * 16;   // this wave's channel tile
    const int p0   = blockIdx.x * 16;             // shared position tile
    const int kbase = (lane >> 4) << 3;           // 0 or 8 (fragment K striping)

    __shared__ _Float16 sB[16][40];               // [pos][k], 16B-aligned rows

    // --- per-thread B-fill coordinates (4 slots, fixed across the K loop) ---
    int fn[4], fk[4], fnB[4], ftB[4];
    bool fok[4];
#pragma unroll
    for (int e = 0; e < 4; ++e) {
        int id = tid + e * 128;
        fn[e] = id >> 5;
        fk[e] = id & 31;
        int pos  = p0 + fn[e];
        int posc = pos < Npos ? pos : Npos - 1;
        fnB[e] = posc / Tout;
        ftB[e] = posc - fnB[e] * Tout;
        fok[e] = pos < Npos;
    }

    // --- A-fragment row (clamped; masked by value, not by exec) ---
    const int  mA   = m0 + (lane & 15);
    const bool mAok = (mA < Cout);
    const int  mAc  = mAok ? mA : Cout - 1;
    const float amsk = mAok ? 1.0f : 0.0f;
    const float* wrow = w + (size_t)mAc * Ktot;

    // --- output coords for this lane ---
    const int  pB   = p0 + (lane & 15);
    const bool pBok = (pB < Npos);
    const int  pBc  = pBok ? pB : Npos - 1;
    const int  nB   = pBc / Tout;
    const int  tB   = pBc - nB * Tout;

    const bool fullK = ((Ktot & 31) == 0);

    v8f acc = {};
    for (int kc = 0; kc < Ktot; kc += 32) {
        __syncthreads();                          // fragments of prev chunk consumed
        // ---- cooperative im2col fill (branch-free clamp+select) ----
#pragma unroll
        for (int e = 0; e < 4; ++e) {
            int kk  = kc + fk[e];
            int kkc = kk < Ktot ? kk : Ktot - 1;
            int ci  = kkc / KS;
            int k   = kkc - ci * KS;
            int tin = ftB[e] * st - pad + k * dil;
            bool ok = fok[e] && (kk < Ktot) && (tin >= 0) && (tin < Tin);
            int  tc = tin < 0 ? 0 : (tin >= Tin ? Tin - 1 : tin);
            float v = x[(fnB[e] * Cin + ci) * Tin + tc];
            v = ok ? v : 0.0f;
            if (relu_in) v = v > 0.0f ? v : 0.0f;
            sB[fn[e]][fk[e]] = (_Float16)v;
        }
        __syncthreads();

        // ---- A fragment ----
        v16h a;
        if (fullK) {
            const v4f* wp = (const v4f*)(wrow + kc);
            const int kb4 = kbase >> 2;
            v4f q0 = wp[kb4], q1 = wp[kb4 + 1], q2 = wp[kb4 + 4], q3 = wp[kb4 + 5];
            __builtin_prefetch(wrow + kc + 32);
#pragma unroll
            for (int i = 0; i < 4; ++i) {
                a[i]      = (_Float16)(q0[i] * amsk);
                a[4 + i]  = (_Float16)(q1[i] * amsk);
                a[8 + i]  = (_Float16)(q2[i] * amsk);
                a[12 + i] = (_Float16)(q3[i] * amsk);
            }
        } else {
#pragma unroll
            for (int i = 0; i < 16; ++i) {
                int kk  = kc + (((i & 8) ? 16 : 0) + kbase + (i & 7));
                int kkc = kk < Ktot ? kk : Ktot - 1;
                float av = wrow[kkc];                 // unconditional (clamped)
                a[i] = (_Float16)((mAok && kk < Ktot) ? av : 0.0f);
            }
        }

        // ---- B fragment from LDS (two ds_load_b128) ----
        const int nn = lane & 15;
        v8h blo = *(const v8h*)&sB[nn][kbase];
        v8h bhi = *(const v8h*)&sB[nn][16 + kbase];
        v16h b;
#pragma unroll
        for (int i = 0; i < 8; ++i) { b[i] = blo[i]; b[8 + i] = bhi[i]; }

        acc = __builtin_amdgcn_wmma_f32_16x16x32_f16(false, a, false, b,
                                                     (short)0, acc, false, false);
    }

#pragma unroll
    for (int r = 0; r < 8; ++r) {
        int m = m0 + r + kbase;
        if (m < Cout && pBok) {
            float v = acc[r] + bias[m];
            if (res) v += res[(nB * Cout + m) * Tout + tB];
            if (relu_out) v = v > 0.0f ? v : 0.0f;
            y[(nB * Cout + m) * Tout + tB] = v;
        }
    }
}

// ---------------------------------------------------------------------------
// WMMA implicit-GEMM 2D convolution (templated on kernel size).
// x: (N, Cin, Hin, Win)  w: (Cout, Cin, KS, KS)  y: (N, Cout, Hout, Wout)
// ---------------------------------------------------------------------------
template <int KS>
__global__ __launch_bounds__(128)
void k_conv2d_wmma(const float* __restrict__ x,
                   const float* __restrict__ w,
                   const float* __restrict__ bias,
                   const float* __restrict__ res,
                   float* __restrict__ y,
                   int N, int Cin, int Hin, int Win,
                   int Cout, int Hout, int Wout,
                   int st, int pad, int dil, int relu_in, int relu_out)
{
    const int KT2  = KS * KS;
    const int Ktot = Cin * KT2;
    const int Npos = N * Hout * Wout;
    const int tid  = threadIdx.x;
    const int lane = tid & 31;
    const int wv   = tid >> 5;
    const int m0   = blockIdx.y * 64 + wv * 16;
    const int p0   = blockIdx.x * 16;
    const int kbase = (lane >> 4) << 3;

    __shared__ _Float16 sB[16][40];

    int fn[4], fk[4], fnB[4], foh[4], fow[4];
    bool fok[4];
#pragma unroll
    for (int e = 0; e < 4; ++e) {
        int id = tid + e * 128;
        fn[e] = id >> 5;
        fk[e] = id & 31;
        int pos  = p0 + fn[e];
        int posc = pos < Npos ? pos : Npos - 1;
        int hw   = Hout * Wout;
        fnB[e] = posc / hw;
        int rem = posc - fnB[e] * hw;
        foh[e] = rem / Wout;
        fow[e] = rem - foh[e] * Wout;
        fok[e] = pos < Npos;
    }

    const int  mA   = m0 + (lane & 15);
    const bool mAok = (mA < Cout);
    const int  mAc  = mAok ? mA : Cout - 1;
    const float amsk = mAok ? 1.0f : 0.0f;
    const float* wrow = w + (size_t)mAc * Ktot;

    const int  pB   = p0 + (lane & 15);
    const bool pBok = (pB < Npos);
    const int  pBc  = pBok ? pB : Npos - 1;
    const int  nB   = pBc / (Hout * Wout);
    const int  remB = pBc - nB * (Hout * Wout);
    const int  oh   = remB / Wout;
    const int  ow   = remB - oh * Wout;

    const bool fullK = ((Ktot & 31) == 0);

    v8f acc = {};
    for (int kc = 0; kc < Ktot; kc += 32) {
        __syncthreads();
#pragma unroll
        for (int e = 0; e < 4; ++e) {
            int kk  = kc + fk[e];
            int kkc = kk < Ktot ? kk : Ktot - 1;
            int ci  = kkc / KT2;
            int r2  = kkc - ci * KT2;
            int kr  = r2 / KS;
            int kc2 = r2 - kr * KS;
            int ih  = foh[e] * st - pad + kr * dil;
            int iw  = fow[e] * st - pad + kc2 * dil;
            bool ok = fok[e] && (kk < Ktot) &&
                      (ih >= 0) && (ih < Hin) && (iw >= 0) && (iw < Win);
            int ihc = ih < 0 ? 0 : (ih >= Hin ? Hin - 1 : ih);
            int iwc = iw < 0 ? 0 : (iw >= Win ? Win - 1 : iw);
            float v = x[((fnB[e] * Cin + ci) * Hin + ihc) * Win + iwc];
            v = ok ? v : 0.0f;
            if (relu_in) v = v > 0.0f ? v : 0.0f;
            sB[fn[e]][fk[e]] = (_Float16)v;
        }
        __syncthreads();

        v16h a;
        if (fullK) {
            const v4f* wp = (const v4f*)(wrow + kc);
            const int kb4 = kbase >> 2;
            v4f q0 = wp[kb4], q1 = wp[kb4 + 1], q2 = wp[kb4 + 4], q3 = wp[kb4 + 5];
            __builtin_prefetch(wrow + kc + 32);
#pragma unroll
            for (int i = 0; i < 4; ++i) {
                a[i]      = (_Float16)(q0[i] * amsk);
                a[4 + i]  = (_Float16)(q1[i] * amsk);
                a[8 + i]  = (_Float16)(q2[i] * amsk);
                a[12 + i] = (_Float16)(q3[i] * amsk);
            }
        } else {
#pragma unroll
            for (int i = 0; i < 16; ++i) {
                int kk  = kc + (((i & 8) ? 16 : 0) + kbase + (i & 7));
                int kkc = kk < Ktot ? kk : Ktot - 1;
                float av = wrow[kkc];                 // unconditional (clamped)
                a[i] = (_Float16)((mAok && kk < Ktot) ? av : 0.0f);
            }
        }

        const int nn = lane & 15;
        v8h blo = *(const v8h*)&sB[nn][kbase];
        v8h bhi = *(const v8h*)&sB[nn][16 + kbase];
        v16h b;
#pragma unroll
        for (int i = 0; i < 8; ++i) { b[i] = blo[i]; b[8 + i] = bhi[i]; }

        acc = __builtin_amdgcn_wmma_f32_16x16x32_f16(false, a, false, b,
                                                     (short)0, acc, false, false);
    }

#pragma unroll
    for (int r = 0; r < 8; ++r) {
        int m = m0 + r + kbase;
        if (m < Cout && pBok) {
            float v = acc[r] + bias[m];
            if (res) v += res[((nB * Cout + m) * Hout + oh) * Wout + ow];
            if (relu_out) v = v > 0.0f ? v : 0.0f;
            y[((nB * Cout + m) * Hout + oh) * Wout + ow] = v;
        }
    }
}

// ---------------------------------------------------------------------------
// WMMA dense layer: y (R,O) = x (R,K) @ w(O,K)^T + bias(O)
// Both operands are K-contiguous -> direct vectorized global loads.
// Remainder path: unconditional clamped loads + value mask (no exec branches).
// ---------------------------------------------------------------------------
__global__ void k_linear_wmma(const float* __restrict__ x,
                              const float* __restrict__ w,
                              const float* __restrict__ bias,
                              float* __restrict__ y,
                              int R, int K, int O)
{
    const int lane = threadIdx.x;
    const int m0 = blockIdx.y * 16;
    const int r0 = blockIdx.x * 16;
    const int kbase = (lane >> 4) << 3;

    const int  mA   = m0 + (lane & 15);
    const bool mAok = (mA < O);
    const float amsk = mAok ? 1.0f : 0.0f;
    const float* wrow = w + (size_t)(mAok ? mA : O - 1) * K;

    const int  rB   = r0 + (lane & 15);
    const bool rBok = (rB < R);
    const float bmsk = rBok ? 1.0f : 0.0f;
    const float* xrow = x + (size_t)(rBok ? rB : R - 1) * K;

    const bool fullK = ((K & 31) == 0);

    v8f acc = {};
    for (int kc = 0; kc < K; kc += 32) {
        v16h a, b;
        if (fullK) {
            const v4f* wp = (const v4f*)(wrow + kc);
            const v4f* xp = (const v4f*)(xrow + kc);
            const int kb4 = kbase >> 2;
            v4f q0 = wp[kb4], q1 = wp[kb4 + 1], q2 = wp[kb4 + 4], q3 = wp[kb4 + 5];
            v4f s0 = xp[kb4], s1 = xp[kb4 + 1], s2 = xp[kb4 + 4], s3 = xp[kb4 + 5];
#pragma unroll
            for (int i = 0; i < 4; ++i) {
                a[i]      = (_Float16)(q0[i] * amsk);
                a[4 + i]  = (_Float16)(q1[i] * amsk);
                a[8 + i]  = (_Float16)(q2[i] * amsk);
                a[12 + i] = (_Float16)(q3[i] * amsk);
                b[i]      = (_Float16)(s0[i] * bmsk);
                b[4 + i]  = (_Float16)(s1[i] * bmsk);
                b[8 + i]  = (_Float16)(s2[i] * bmsk);
                b[12 + i] = (_Float16)(s3[i] * bmsk);
            }
        } else {
#pragma unroll
            for (int i = 0; i < 16; ++i) {
                int kk  = kc + (((i & 8) ? 16 : 0) + kbase + (i & 7));
                int kkc = kk < K ? kk : K - 1;
                bool kok = kk < K;
                float av = wrow[kkc];                 // unconditional (clamped)
                float bv = xrow[kkc];                 // unconditional (clamped)
                a[i] = (_Float16)((mAok && kok) ? av : 0.0f);
                b[i] = (_Float16)((rBok && kok) ? bv : 0.0f);
            }
        }
        acc = __builtin_amdgcn_wmma_f32_16x16x32_f16(false, a, false, b,
                                                     (short)0, acc, false, false);
    }
#pragma unroll
    for (int r = 0; r < 8; ++r) {
        int m = m0 + r + kbase;
        if (m < O && rBok) y[(size_t)rB * O + m] = acc[r] + bias[m];
    }
}

// ---------------------------------------------------------------------------
// Residual VQ: one 256-thread block per row. C=256, NB=256, NQ=6.
// ---------------------------------------------------------------------------
__global__ void k_rvq(const float* __restrict__ flat,
                      const float* __restrict__ cb,
                      float* __restrict__ quant,
                      float* __restrict__ lossacc, int R)
{
    __shared__ float sres[256];
    __shared__ float rd[256];
    __shared__ int   ri[256];
    const int r = blockIdx.x;
    const int tid = threadIdx.x;
    if (r >= R) return;
    sres[tid] = flat[r * 256 + tid];
    float qacc = 0.0f;
    for (int q = 0; q < 6; ++q) {
        __syncthreads();
        const float* cbq = cb + ((q * 256 + tid) * 256);
        float d = 0.0f;
        for (int c = 0; c < 256; ++c) {
            float t = sres[c] - cbq[c];
            d += t * t;
        }
        rd[tid] = d; ri[tid] = tid;
        __syncthreads();
        for (int s = 128; s > 0; s >>= 1) {
            if (tid < s) {
                float dv = rd[tid + s]; int iv = ri[tid + s];
                if (dv < rd[tid] || (dv == rd[tid] && iv < ri[tid])) {
                    rd[tid] = dv; ri[tid] = iv;
                }
            }
            __syncthreads();
        }
        int best = ri[0];
        float sel = cb[(q * 256 + best) * 256 + tid];
        float nres = sres[tid] - sel;
        qacc += sel;
        __syncthreads();
        rd[tid] = nres * nres;
        __syncthreads();
        for (int s = 128; s > 0; s >>= 1) {
            if (tid < s) rd[tid] += rd[tid + s];
            __syncthreads();
        }
        if (tid == 0) atomicAdd(&lossacc[q], rd[0]);
        sres[tid] = nres;
    }
    quant[r * 256 + tid] = qacc;
}

// ------------------------ elementwise / layout kernels ----------------------
__global__ void k_zero(float* p, int n) {
    int i = blockIdx.x * blockDim.x + threadIdx.x;
    if (i < n) p[i] = 0.0f;
}
__global__ void k_ntc_to_nct(const float* in, float* out, int N, int T, int C) {
    int idx = blockIdx.x * blockDim.x + threadIdx.x;
    int total = N * T * C;
    if (idx >= total) return;
    int c = idx % C; int t = (idx / C) % T; int n = idx / (C * T);
    out[(n * C + c) * T + t] = in[idx];
}
__global__ void k_nct_to_ntc(const float* in, float* out, int N, int C, int T) {
    int idx = blockIdx.x * blockDim.x + threadIdx.x;
    int total = N * C * T;
    if (idx >= total) return;
    int t = idx % T; int c = (idx / T) % C; int n = idx / (T * C);
    out[(n * T + t) * C + c] = in[idx];
}
__global__ void k_up1d(const float* in, float* out, int N, int C, int Tin) {
    int Tout = Tin * 2;
    int idx = blockIdx.x * blockDim.x + threadIdx.x;
    int total = N * C * Tout;
    if (idx >= total) return;
    int t = idx % Tout; int c = (idx / Tout) % C; int n = idx / (Tout * C);
    out[idx] = in[(n * C + c) * Tin + (t >> 1)];
}
__global__ void k_up2d(const float* in, float* out, int N, int C, int Hin, int Win) {
    int Hout = Hin * 2, Wout = Win * 2;
    int idx = blockIdx.x * blockDim.x + threadIdx.x;
    int total = N * C * Hout * Wout;
    if (idx >= total) return;
    int wI = idx % Wout; int h = (idx / Wout) % Hout;
    int c = (idx / (Wout * Hout)) % C; int n = idx / (Wout * Hout * C);
    out[idx] = in[((n * C + c) * Hin + (h >> 1)) * Win + (wI >> 1)];
}
__global__ void k_x2prep(const float* x2d, float* out) {
    int idx = blockIdx.x * blockDim.x + threadIdx.x;
    const int total = 8 * 6 * 256 * 32;
    if (idx >= total) return;
    int j = idx % 32; int t = (idx / 32) % 256;
    int d = (idx / (32 * 256)) % 6; int n = idx / (32 * 256 * 6);
    float v = 0.0f;
    if (j >= 1 && j <= 30) v = x2d[((n * 256 + t) * 30 + (j - 1)) * 6 + d];
    out[idx] = v;
}
__global__ void k_xe2_to_flat(const float* xe2, float* flat) {
    int idx = blockIdx.x * blockDim.x + threadIdx.x;
    const int total = 8 * 4 * 32 * 256;
    if (idx >= total) return;
    int d = idx % 256; int t = (idx / 256) % 32;
    int j = (idx / (256 * 32)) % 4; int b = idx / (256 * 32 * 4);
    flat[idx] = xe2[((b * 256 + d) * 32 + t) * 4 + j];
}
__global__ void k_q2_to_min(const float* q2, float* minb) {
    int idx = blockIdx.x * blockDim.x + threadIdx.x;
    const int total = 256 * 1024;
    if (idx >= total) return;
    int col = idx % 1024; int row = idx / 1024;
    int b = row / 32, t = row % 32;
    int j = col / 256, d = col % 256;
    minb[idx] = q2[((b * 4 + j) * 32 + t) * 256 + d];
}
__global__ void k_min_to_decin(const float* mo, float* dec) {
    int idx = blockIdx.x * blockDim.x + threadIdx.x;
    const int total = 8 * 256 * 32 * 4;
    if (idx >= total) return;
    int j = idx % 4; int t = (idx / 4) % 32;
    int d = (idx / (4 * 32)) % 256; int b = idx / (4 * 32 * 256);
    dec[idx] = mo[(b * 32 + t) * 1024 + j * 256 + d];
}
__global__ void k_out2(const float* y, float* out2) {
    int idx = blockIdx.x * blockDim.x + threadIdx.x;
    const int total = 8 * 256 * 180;
    if (idx >= total) return;
    int c = idx % 180; int t = (idx / 180) % 256; int b = idx / (180 * 256);
    int j = c / 6; int d = c % 6;
    out2[idx] = y[((b * 6 + d) * 256 + t) * 32 + (j + 1)];
}
__global__ void k_cat(const float* a, const float* b2, float* cat) {
    int idx = blockIdx.x * blockDim.x + threadIdx.x;
    const int total = 2048 * 360;
    if (idx >= total) return;
    int c = idx % 360; int r = idx / 360;
    cat[idx] = (c < 180) ? a[r * 180 + c] : b2[r * 180 + (c - 180)];
}
__global__ void k_loss_final(const float* acc, float* out) {
    int i = threadIdx.x;
    if (i < 6)       out[i] = acc[i] / (512.0f * 256.0f);
    else if (i < 12) out[i] = acc[i] / (1024.0f * 256.0f);
}

// ---------------------------- host orchestration ----------------------------
static inline void conv1d_l(hipStream_t s, const float* x, const float* w, const float* b,
                            const float* res, float* y, int N, int Cin, int Tin,
                            int Cout, int Tout, int ks, int st, int pad, int dil,
                            int ri, int ro)
{
    dim3 g((unsigned)cdiv(N * Tout, 16), (unsigned)cdiv(Cout, 64));
    dim3 blk(128);
    switch (ks) {
    case 1: k_conv1d_wmma<1><<<g, blk, 0, s>>>(x, w, b, res, y, N, Cin, Tin, Cout, Tout, st, pad, dil, ri, ro); break;
    case 3: k_conv1d_wmma<3><<<g, blk, 0, s>>>(x, w, b, res, y, N, Cin, Tin, Cout, Tout, st, pad, dil, ri, ro); break;
    case 4: k_conv1d_wmma<4><<<g, blk, 0, s>>>(x, w, b, res, y, N, Cin, Tin, Cout, Tout, st, pad, dil, ri, ro); break;
    default: break;
    }
}
static inline void conv2d_l(hipStream_t s, const float* x, const float* w, const float* b,
                            const float* res, float* y, int N, int Cin, int Hin, int Win,
                            int Cout, int Hout, int Wout, int ks, int st, int pad, int dil,
                            int ri, int ro)
{
    dim3 g((unsigned)cdiv(N * Hout * Wout, 16), (unsigned)cdiv(Cout, 64));
    dim3 blk(128);
    switch (ks) {
    case 1: k_conv2d_wmma<1><<<g, blk, 0, s>>>(x, w, b, res, y, N, Cin, Hin, Win, Cout, Hout, Wout, st, pad, dil, ri, ro); break;
    case 3: k_conv2d_wmma<3><<<g, blk, 0, s>>>(x, w, b, res, y, N, Cin, Hin, Win, Cout, Hout, Wout, st, pad, dil, ri, ro); break;
    case 4: k_conv2d_wmma<4><<<g, blk, 0, s>>>(x, w, b, res, y, N, Cin, Hin, Win, Cout, Hout, Wout, st, pad, dil, ri, ro); break;
    default: break;
    }
}
static inline float* other3(float* b0, float* b1, float* b2, const float* x, const float* y) {
    if (b0 != x && b0 != y) return b0;
    if (b1 != x && b1 != y) return b1;
    return b2;
}
#define EW(kern, total, ...) \
    kern<<<dim3((unsigned)cdiv((total), 256)), dim3(256), 0, stream>>>(__VA_ARGS__)

extern "C" void kernel_launch(void* const* d_in, const int* in_sizes, int n_in,
                              void* d_out_v, int out_size, void* d_ws, size_t ws_size,
                              hipStream_t stream)
{
    (void)in_sizes; (void)n_in; (void)out_size; (void)ws_size;
    const float* x1d = (const float*)d_in[0];
    const float* x2d = (const float*)d_in[1];
    auto P = [&](int i) -> const float* { return (const float*)d_in[2 + i]; };
    float* out = (float*)d_out_v;
    float* ws  = (float*)d_ws;

    const size_t BUFN = 16777216;           // (8,256,256,32) f32
    float* b0 = ws;
    float* b1 = ws + BUFN;
    float* b2 = ws + 2 * BUFN;
    float* sm = ws + 3 * BUFN;
    float* flat1  = sm;
    float* quant1 = flat1 + 131072;
    float* flat2  = quant1 + 131072;
    float* quant2 = flat2 + 262144;
    float* minb   = quant2 + 262144;
    float* mob    = minb + 262144;
    float* out2   = mob + 262144;
    float* cat    = out2 + 368640;
    float* lossa  = cat + 737280;

    float* out1d   = out;
    float* rec2d   = out + 368640;
    float* lossout = out + 737280;

    const int dilF[3] = {1, 3, 9};
    const int dilR[3] = {9, 3, 1};

    EW(k_zero, 12, lossa, 12);

    // ============================= 1D branch ================================
    EW(k_ntc_to_nct, 8 * 256 * 180, x1d, b0, 8, 256, 180);
    conv1d_l(stream, b0, P(0), P(1), nullptr, b1, 8, 180, 256, 256, 256, 3, 1, 1, 1, 0, 1);
    float* cur = b1;
    int T = 256;
    for (int L = 0; L < 2; ++L) {
        int base = 2 + L * 14;
        int Tn = T / 2;
        float* dst = other3(b0, b1, b2, cur, nullptr);
        conv1d_l(stream, cur, P(base), P(base + 1), nullptr, dst,
                 8, 256, T, 256, Tn, 4, 2, 1, 1, 0, 0);
        cur = dst; T = Tn;
        float* X = cur;
        for (int k = 0; k < 3; ++k) {
            int bb = base + 2 + k * 4;
            int d = dilF[k];
            float* Hb = other3(b0, b1, b2, X, nullptr);
            float* Ob = other3(b0, b1, b2, X, Hb);
            conv1d_l(stream, X, P(bb), P(bb + 1), nullptr, Hb, 8, 256, T, 256, T, 3, 1, d, d, 1, 0);
            conv1d_l(stream, Hb, P(bb + 2), P(bb + 3), X, Ob, 8, 256, T, 256, T, 1, 1, 0, 1, 1, 0);
            X = Ob;
        }
        cur = X;
    }
    {
        float* dst = other3(b0, b1, b2, cur, nullptr);
        conv1d_l(stream, cur, P(30), P(31), nullptr, dst, 8, 256, 64, 256, 64, 3, 1, 1, 1, 0, 0);
        cur = dst;                                      // xe1 (8,256,64)
    }
    EW(k_nct_to_ntc, 8 * 256 * 64, cur, flat1, 8, 256, 64);
    k_rvq<<<dim3(512), dim3(256), 0, stream>>>(flat1, P(160), quant1, lossa, 512);
    EW(k_ntc_to_nct, 512 * 256, quant1, b0, 8, 64, 256); // xq1 -> (8,256,64)

    // decoder1d
    conv1d_l(stream, b0, P(32), P(33), nullptr, b1, 8, 256, 64, 256, 64, 3, 1, 1, 1, 0, 1);
    cur = b1; T = 64;
    for (int L = 0; L < 2; ++L) {
        int lb = 34 + L * 14;
        float* X = cur;
        for (int k = 0; k < 3; ++k) {
            int bb = lb + k * 4;
            int d = dilR[k];
            float* Hb = other3(b0, b1, b2, X, nullptr);
            float* Ob = other3(b0, b1, b2, X, Hb);
            conv1d_l(stream, X, P(bb), P(bb + 1), nullptr, Hb, 8, 256, T, 256, T, 3, 1, d, d, 1, 0);
            conv1d_l(stream, Hb, P(bb + 2), P(bb + 3), X, Ob, 8, 256, T, 256, T, 1, 1, 0, 1, 1, 0);
            X = Ob;
        }
        cur = X;
        float* up = other3(b0, b1, b2, cur, nullptr);
        EW(k_up1d, 8 * 256 * (T * 2), cur, up, 8, 256, T);
        T *= 2;
        float* dst = other3(b0, b1, b2, up, nullptr);
        conv1d_l(stream, up, P(lb + 12), P(lb + 13), nullptr, dst, 8, 256, T, 256, T, 3, 1, 1, 1, 0, 0);
        cur = dst;
    }
    {
        float* dst = other3(b0, b1, b2, cur, nullptr);
        conv1d_l(stream, cur, P(62), P(63), nullptr, dst, 8, 256, 256, 256, 256, 3, 1, 1, 1, 0, 1);
        cur = dst;
        dst = other3(b0, b1, b2, cur, nullptr);
        conv1d_l(stream, cur, P(64), P(65), nullptr, dst, 8, 256, 256, 180, 256, 3, 1, 1, 1, 0, 0);
        cur = dst;
    }
    EW(k_nct_to_ntc, 8 * 180 * 256, cur, out1d, 8, 180, 256);

    // ============================= 2D branch ================================
    EW(k_x2prep, 8 * 6 * 256 * 32, x2d, b0);
    conv2d_l(stream, b0, P(66), P(67), nullptr, b1, 8, 6, 256, 32, 256, 256, 32, 3, 1, 1, 1, 0, 1);
    cur = b1; int Hh = 256, Wd = 32;
    for (int L = 0; L < 3; ++L) {
        int lb = 68 + L * 14;
        int Hn = Hh / 2, Wn = Wd / 2;
        float* dst = other3(b0, b1, b2, cur, nullptr);
        conv2d_l(stream, cur, P(lb), P(lb + 1), nullptr, dst,
                 8, 256, Hh, Wd, 256, Hn, Wn, 4, 2, 1, 1, 0, 0);
        cur = dst; Hh = Hn; Wd = Wn;
        float* X = cur;
        for (int k = 0; k < 3; ++k) {
            int bb = lb + 2 + k * 4;
            int d = dilF[k];
            float* Hb = other3(b0, b1, b2, X, nullptr);
            float* Ob = other3(b0, b1, b2, X, Hb);
            conv2d_l(stream, X, P(bb), P(bb + 1), nullptr, Hb,
                     8, 256, Hh, Wd, 256, Hh, Wd, 3, 1, d, d, 1, 0);
            conv2d_l(stream, Hb, P(bb + 2), P(bb + 3), X, Ob,
                     8, 256, Hh, Wd, 256, Hh, Wd, 1, 1, 0, 1, 1, 0);
            X = Ob;
        }
        cur = X;
    }
    {
        float* dst = other3(b0, b1, b2, cur, nullptr);
        conv2d_l(stream, cur, P(110), P(111), nullptr, dst, 8, 256, 32, 4, 256, 32, 4, 3, 1, 1, 1, 0, 0);
        cur = dst;                                      // xe2 (8,256,32,4)
    }
    EW(k_xe2_to_flat, 8 * 4 * 32 * 256, cur, flat2);
    k_rvq<<<dim3(1024), dim3(256), 0, stream>>>(flat2, P(161), quant2, lossa + 6, 1024);
    EW(k_q2_to_min, 256 * 1024, quant2, minb);
    {
        dim3 g((unsigned)cdiv(256, 16), (unsigned)cdiv(1024, 16));
        k_linear_wmma<<<g, dim3(32), 0, stream>>>(minb, P(162), P(163), mob, 256, 1024, 1024);
    }
    EW(k_min_to_decin, 8 * 256 * 32 * 4, mob, b0);

    // decoder2d
    conv2d_l(stream, b0, P(112), P(113), nullptr, b1, 8, 256, 32, 4, 256, 32, 4, 3, 1, 1, 1, 0, 1);
    cur = b1; Hh = 32; Wd = 4;
    for (int L = 0; L < 3; ++L) {
        int lb = 114 + L * 14;
        float* X = cur;
        for (int k = 0; k < 3; ++k) {
            int bb = lb + k * 4;
            int d = dilR[k];
            float* Hb = other3(b0, b1, b2, X, nullptr);
            float* Ob = other3(b0, b1, b2, X, Hb);
            conv2d_l(stream, X, P(bb), P(bb + 1), nullptr, Hb,
                     8, 256, Hh, Wd, 256, Hh, Wd, 3, 1, d, d, 1, 0);
            conv2d_l(stream, Hb, P(bb + 2), P(bb + 3), X, Ob,
                     8, 256, Hh, Wd, 256, Hh, Wd, 1, 1, 0, 1, 1, 0);
            X = Ob;
        }
        cur = X;
        float* up = other3(b0, b1, b2, cur, nullptr);
        EW(k_up2d, 8 * 256 * (Hh * 2) * (Wd * 2), cur, up, 8, 256, Hh, Wd);
        Hh *= 2; Wd *= 2;
        float* dst = other3(b0, b1, b2, up, nullptr);
        conv2d_l(stream, up, P(lb + 12), P(lb + 13), nullptr, dst,
                 8, 256, Hh, Wd, 256, Hh, Wd, 3, 1, 1, 1, 0, 0);
        cur = dst;
    }
    {
        float* dst = other3(b0, b1, b2, cur, nullptr);
        conv2d_l(stream, cur, P(156), P(157), nullptr, dst, 8, 256, 256, 32, 256, 256, 32, 3, 1, 1, 1, 0, 1);
        cur = dst;
        dst = other3(b0, b1, b2, cur, nullptr);
        conv2d_l(stream, cur, P(158), P(159), nullptr, dst, 8, 256, 256, 32, 6, 256, 32, 3, 1, 1, 1, 0, 0);
        cur = dst;                                      // (8,6,256,32)
    }
    EW(k_out2, 8 * 256 * 180, cur, out2);

    // ====================== final projection + losses =======================
    EW(k_cat, 2048 * 360, out1d, out2, cat);
    {
        dim3 g((unsigned)cdiv(2048, 16), (unsigned)cdiv(180, 16));
        k_linear_wmma<<<g, dim3(32), 0, stream>>>(cat, P(164), P(165), rec2d, 2048, 360, 180);
    }
    k_loss_final<<<dim3(1), dim3(12), 0, stream>>>(lossa, lossout);
}
